// KnowledgeAdapterBlock_75376676045570
// MI455X (gfx1250) — compile-verified, matrix operand order checked by reading, not run on hardware
//
#include <hip/hip_runtime.h>

// ---------------------------------------------------------------------------
// MI455X (gfx1250, wave32) implementation of the KnowledgeAdapter block.
// All GEMMs run on v_wmma_f32_16x16x32_f16 (f16 A/B, f32 accum).
// Both attentions are fused (WMMA scores -> LDS softmax -> WMMA PV).
// GEMM wave tile is 32x64 (8 WMMA per 32-K step) to maximize the
// wmma:vmem issue ratio on the compute-bound matmuls.
// ---------------------------------------------------------------------------

typedef _Float16 f16;
typedef __attribute__((ext_vector_type(16))) _Float16 v16h;
typedef __attribute__((ext_vector_type(8)))  float    v8f;

#define DEV __device__ __forceinline__

enum { GF_OUTF16 = 1, GF_BIAS = 2, GF_ADD = 4, GF_GELU = 8, GF_REMAP = 16 };

// ---- fragment helpers ------------------------------------------------------
// 16-bit WMMA fragments as loaded here (contiguous K; AMD's table order differs): both A (16x32) and B (32x16, fed as rows of
// B^T) are "16 contiguous K-halves per lane": lanes 0-15 take K 0-15,
// lanes 16-31 take K 16-31, lane&15 selects the M (or N) row.

DEV v16h ld16(const f16* p) {
  union { uint4 u[2]; v16h v; } r;
  r.u[0] = *(const uint4*)(p);
  r.u[1] = *(const uint4*)(p + 8);
  return r.v;
}

DEV v16h ldfrag(const f16* base, long rowStride, int r0, int k0) {
  const int lane = threadIdx.x & 31;
  return ld16(base + (long)(r0 + (lane & 15)) * rowStride + k0 + ((lane >> 4) << 4));
}

DEV v8f wmma32(v16h a, v16h b, v8f c) {
  return __builtin_amdgcn_wmma_f32_16x16x32_f16(false, a, false, b, (short)0, c,
                                                false, false);
}

// C/D fragment mapping: VGPR r of lane l holds M = r + 8*(l>=16), N = l&15.

// ---- generic WMMA GEMM: C[M,N] = A[M,K] @ BT[N,K]^T  (epilogue flags) ------

DEV void gemm_store_tile(v8f a, int row0, int col0, int N,
                         const float* __restrict__ bias,
                         const float* __restrict__ Cin,
                         void* __restrict__ Cout, int flags) {
  const int lane = threadIdx.x & 31;
  const int mb = (lane >> 4) << 3;
  const int nn = lane & 15;
#pragma unroll
  for (int r = 0; r < 8; ++r) {
    const int row = row0 + mb + r;
    const int col = col0 + nn;
    float v = a[r];
    if (flags & GF_BIAS) v += bias[col];
    if (flags & GF_GELU) v = v / (1.0f + __expf(-1.702f * v));  // QuickGELU
    const long orow = (flags & GF_REMAP)
                          ? (long)((row & 63) * 128 + (row >> 6))  // (bt,l)->(l,bt)
                          : (long)row;
    if (flags & GF_ADD) v += Cin[orow * (long)N + col];
    if (flags & GF_OUTF16) ((f16*)Cout)[orow * (long)N + col] = (f16)v;
    else                   ((float*)Cout)[orow * (long)N + col] = v;
  }
}

// Block tile 64x128 (4 waves); each wave computes a 32x64 tile:
// per 32-K step: 2 A frags + 4 B frags feed 8 WMMAs.
__global__ __launch_bounds__(128)
void gemm_wmma_kernel(const f16* __restrict__ A, long lda,
                      const f16* __restrict__ BT, long ldb,
                      const float* __restrict__ bias,
                      const float* __restrict__ Cin,
                      void* __restrict__ Cout,
                      int N, int K, int flags) {
  const int wave = threadIdx.x >> 5;
  const int m0 = blockIdx.y * 64 + (wave & 1) * 32;
  const int n0 = blockIdx.x * 128 + (wave >> 1) * 64;
  v8f acc[2][4];
#pragma unroll
  for (int i = 0; i < 2; ++i)
#pragma unroll
    for (int j = 0; j < 4; ++j) acc[i][j] = (v8f){};

  for (int k = 0; k < K; k += 32) {
    const v16h a0 = ldfrag(A, lda, m0, k);
    const v16h a1 = ldfrag(A, lda, m0 + 16, k);
#pragma unroll
    for (int j = 0; j < 4; ++j) {
      const v16h b = ldfrag(BT, ldb, n0 + j * 16, k);
      acc[0][j] = wmma32(a0, b, acc[0][j]);
      acc[1][j] = wmma32(a1, b, acc[1][j]);
    }
  }
#pragma unroll
  for (int i = 0; i < 2; ++i)
#pragma unroll
    for (int j = 0; j < 4; ++j)
      gemm_store_tile(acc[i][j], m0 + i * 16, n0 + j * 16, N, bias, Cin, Cout,
                      flags);
}

// ---- LayerNorm over D=768, optional second LN fused, f16 output ------------

DEV float block_sum_768(float v, float* red) {
  const int t = threadIdx.x;
  red[t] = v;
  __syncthreads();
  for (int o = 128; o > 0; o >>= 1) {
    if (t < o) red[t] += red[t + o];
    __syncthreads();
  }
  const float r = red[0];
  __syncthreads();
  return r;
}

__global__ __launch_bounds__(256)
void ln768_kernel(const float* __restrict__ in, f16* __restrict__ out,
                  const float* __restrict__ g1, const float* __restrict__ b1,
                  const float* __restrict__ g2, const float* __restrict__ b2) {
  __shared__ float red[256];
  const int t = threadIdx.x;
  const long row = blockIdx.x;
  const float* x = in + row * 768;
  float v0 = x[t], v1 = x[t + 256], v2 = x[t + 512];
  const float mean = block_sum_768(v0 + v1 + v2, red) * (1.0f / 768.0f);
  float d0 = v0 - mean, d1 = v1 - mean, d2 = v2 - mean;
  const float var = block_sum_768(d0 * d0 + d1 * d1 + d2 * d2, red) * (1.0f / 768.0f);
  const float rs = rsqrtf(var + 1e-5f);
  float y0 = d0 * rs * g1[t] + b1[t];
  float y1 = d1 * rs * g1[t + 256] + b1[t + 256];
  float y2 = d2 * rs * g1[t + 512] + b1[t + 512];
  if (g2 != nullptr) {  // fused second LN (norm_latents -> norm_latents2)
    const float m2 = block_sum_768(y0 + y1 + y2, red) * (1.0f / 768.0f);
    float e0 = y0 - m2, e1 = y1 - m2, e2 = y2 - m2;
    const float vv = block_sum_768(e0 * e0 + e1 * e1 + e2 * e2, red) * (1.0f / 768.0f);
    const float r2 = rsqrtf(vv + 1e-5f);
    y0 = e0 * r2 * g2[t] + b2[t];
    y1 = e1 * r2 * g2[t + 256] + b2[t + 256];
    y2 = e2 * r2 * g2[t + 512] + b2[t + 512];
  }
  f16* o = out + row * 768;
  o[t] = (f16)y0;
  o[t + 256] = (f16)y1;
  o[t + 512] = (f16)y2;
}

// ---- weight transpose + f16 pack:  out[N,K] = (f16) in[K,N] ----------------

__global__ __launch_bounds__(256)
void transpose_f32_f16(const float* __restrict__ in, f16* __restrict__ out,
                       int R, int C) {
  const long idx = (long)blockIdx.x * 256 + threadIdx.x;
  if (idx >= (long)R * C) return;
  const int r = (int)(idx / C);
  const int c = (int)(idx % C);
  out[(long)c * R + r] = (f16)in[idx];
}

// ---- fused attention helpers ----------------------------------------------

DEV void store_scores(f16* S, int ldS, int m0, int col0, v8f a, float scale) {
  const int lane = threadIdx.x & 31;
  const int mb = (lane >> 4) << 3;
  const int nn = lane & 15;
#pragma unroll
  for (int r = 0; r < 8; ++r)
    S[(m0 + mb + r) * ldS + col0 + nn] = (f16)(a[r] * scale);
}

DEV void softmax_row(f16* r, int n) {
  float mx = -3.0e38f;
  for (int j = 0; j < n; ++j) mx = fmaxf(mx, (float)r[j]);
  float sum = 0.0f;
  for (int j = 0; j < n; ++j) {
    const float e = __expf((float)r[j] - mx);
    sum += e;
    r[j] = (f16)e;
  }
  const float inv = 1.0f / sum;
  for (int j = 0; j < n; ++j) r[j] = (f16)((float)r[j] * inv);
}

// key fragment source for Perceiver attention: keys j<256 come from
// image kv (n, bt, 1024); j>=256 from latent kv (t*64+l, 1024).
DEV v16h ld_key_frag(const f16* __restrict__ kvx, const f16* __restrict__ kvl,
                     int bt, int t, int h, int j0, int k0) {
  const int lane = threadIdx.x & 31;
  const int j = j0 + (lane & 15);
  const int cc = h * 64 + k0 + ((lane >> 4) << 4);
  const f16* p = (j < 256)
                     ? kvx + (long)j * 131072 + (long)bt * 1024 + cc
                     : kvl + (long)(t * 64 + (j - 256)) * 1024 + cc;
  return ld16(p);
}

// ---- Perceiver cross-attention: 1 block per (bt, head), 4 waves ------------
// S(64x320) = q(64x64) @ K^T, softmax, O(64x64) = P @ V.  S kept in LDS (f16),
// V^T staged through LDS in 32-key chunks.

__global__ __launch_bounds__(128)
void perceiver_attn_kernel(const f16* __restrict__ qf,
                           const f16* __restrict__ kvx,
                           const f16* __restrict__ kvl,
                           f16* __restrict__ Oout) {
  __shared__ f16 S[64 * 320];   // 40 KB
  __shared__ f16 VT[64 * 32];   // 4 KB  (dh-major transposed V chunk)
  const int bt = blockIdx.x >> 3;
  const int h = blockIdx.x & 7;
  const int t = bt & 7;  // bt = b*8 + t
  const int tid = threadIdx.x;
  const int wave = tid >> 5;
  const int lane = tid & 31;
  const int m0 = wave * 16;

  // q rows are (t*64+l); scale 1/sqrt(64) folded into score store
  const f16* qb = qf + (long)(t * 64) * 512 + h * 64;
  const v16h a0 = ldfrag(qb, 512, m0, 0);
  const v16h a1 = ldfrag(qb, 512, m0, 32);

  for (int nc = 0; nc < 10; ++nc) {  // 320 keys in chunks of 32
    const int j0 = nc * 32;
    v8f s0 = {}, s1 = {};
    s0 = wmma32(a0, ld_key_frag(kvx, kvl, bt, t, h, j0, 0), s0);
    s0 = wmma32(a1, ld_key_frag(kvx, kvl, bt, t, h, j0, 32), s0);
    s1 = wmma32(a0, ld_key_frag(kvx, kvl, bt, t, h, j0 + 16, 0), s1);
    s1 = wmma32(a1, ld_key_frag(kvx, kvl, bt, t, h, j0 + 16, 32), s1);
    store_scores(S, 320, m0, j0, s0, 0.125f);
    store_scores(S, 320, m0, j0 + 16, s1, 0.125f);
  }
  __syncthreads();
  if (tid < 64) softmax_row(S + tid * 320, 320);
  __syncthreads();

  v8f o0 = {}, o1 = {}, o2 = {}, o3 = {};
  for (int kc = 0; kc < 10; ++kc) {
    const int j0 = kc * 32;
    for (int e = tid; e < 2048; e += 128) {  // stage V^T chunk: VT[dh][jj]
      const int dh = e & 63, jj = e >> 6;
      const int j = j0 + jj;
      const int cc = 512 + h * 64 + dh;  // v half of kv
      VT[dh * 32 + jj] =
          (j < 256) ? kvx[(long)j * 131072 + (long)bt * 1024 + cc]
                    : kvl[(long)(t * 64 + (j - 256)) * 1024 + cc];
    }
    __syncthreads();
    const v16h p = ldfrag(S, 320, m0, j0);
    o0 = wmma32(p, ldfrag(VT, 32, 0, 0), o0);
    o1 = wmma32(p, ldfrag(VT, 32, 16, 0), o1);
    o2 = wmma32(p, ldfrag(VT, 32, 32, 0), o2);
    o3 = wmma32(p, ldfrag(VT, 32, 48, 0), o3);
    __syncthreads();
  }
  // store O rows (bt*64 + l), cols h*64+dh  -> layout (b,t,L,INNER)
  const int mb = (lane >> 4) << 3;
  const int nn = lane & 15;
#pragma unroll
  for (int r = 0; r < 8; ++r) {
    const long row = (long)bt * 64 + m0 + mb + r;
    f16* dst = Oout + row * 512 + h * 64 + nn;
    dst[0] = (f16)o0[r];
    dst[16] = (f16)o1[r];
    dst[32] = (f16)o2[r];
    dst[48] = (f16)o3[r];
  }
}

// ---- self-attention: 1 block per (bt, head12), seq = 64 --------------------

__global__ __launch_bounds__(128)
void self_attn_kernel(const f16* __restrict__ qkv, f16* __restrict__ Oout) {
  __shared__ f16 S[64 * 64];
  __shared__ f16 VT[64 * 64];
  const int bt = blockIdx.x / 12;
  const int h = blockIdx.x % 12;
  const int tid = threadIdx.x;
  const int wave = tid >> 5;
  const int lane = tid & 31;
  const int m0 = wave * 16;
  const long rs = 128L * 2304;  // seq-row stride of qkv (L, bt, 2304)

  const f16* qb = qkv + (long)bt * 2304 + h * 64;
  const f16* kb = qb + 768;
  const f16* vb = qb + 1536;
  const v16h a0 = ldfrag(qb, rs, m0, 0);
  const v16h a1 = ldfrag(qb, rs, m0, 32);
#pragma unroll
  for (int nc = 0; nc < 2; ++nc) {
    const int j0 = nc * 32;
    v8f s0 = {}, s1 = {};
    s0 = wmma32(a0, ldfrag(kb, rs, j0, 0), s0);
    s0 = wmma32(a1, ldfrag(kb, rs, j0, 32), s0);
    s1 = wmma32(a0, ldfrag(kb, rs, j0 + 16, 0), s1);
    s1 = wmma32(a1, ldfrag(kb, rs, j0 + 16, 32), s1);
    store_scores(S, 64, m0, j0, s0, 0.125f);
    store_scores(S, 64, m0, j0 + 16, s1, 0.125f);
  }
  for (int e = tid; e < 4096; e += 128) {  // stage full V^T
    const int dh = e & 63, jj = e >> 6;
    VT[dh * 64 + jj] = vb[(long)jj * rs + dh];
  }
  __syncthreads();
  if (tid < 64) softmax_row(S + tid * 64, 64);
  __syncthreads();

  v8f o0 = {}, o1 = {}, o2 = {}, o3 = {};
#pragma unroll
  for (int kc = 0; kc < 2; ++kc) {
    const v16h p = ldfrag(S, 64, m0, kc * 32);
    o0 = wmma32(p, ldfrag(VT, 64, 0, kc * 32), o0);
    o1 = wmma32(p, ldfrag(VT, 64, 16, kc * 32), o1);
    o2 = wmma32(p, ldfrag(VT, 64, 32, kc * 32), o2);
    o3 = wmma32(p, ldfrag(VT, 64, 48, kc * 32), o3);
  }
  const int mb = (lane >> 4) << 3;
  const int nn = lane & 15;
#pragma unroll
  for (int r = 0; r < 8; ++r) {
    const long l = m0 + mb + r;
    f16* dst = Oout + (l * 128 + bt) * 768 + h * 64 + nn;
    dst[0] = (f16)o0[r];
    dst[16] = (f16)o1[r];
    dst[32] = (f16)o2[r];
    dst[48] = (f16)o3[r];
  }
}

// ---------------------------------------------------------------------------

extern "C" void kernel_launch(void* const* d_in, const int* in_sizes, int n_in,
                              void* d_out, int out_size, void* d_ws,
                              size_t ws_size, hipStream_t stream) {
  (void)in_sizes; (void)n_in; (void)out_size; (void)ws_size;
  const float* visual = (const float*)d_in[0];
  const float* latents = (const float*)d_in[1];
  const float* nl0_g = (const float*)d_in[2];
  const float* nl0_b = (const float*)d_in[3];
  const float* pm_g = (const float*)d_in[4];
  const float* pm_b = (const float*)d_in[5];
  const float* pl_g = (const float*)d_in[6];
  const float* pl_b = (const float*)d_in[7];
  const float* Wq = (const float*)d_in[8];
  const float* Wkv = (const float*)d_in[9];
  const float* Wo = (const float*)d_in[10];
  const float* ln1_g = (const float*)d_in[11];
  const float* ln1_b = (const float*)d_in[12];
  const float* in_proj_w = (const float*)d_in[13];
  const float* in_proj_b = (const float*)d_in[14];
  const float* out_w = (const float*)d_in[15];
  const float* out_b = (const float*)d_in[16];
  const float* ln2_g = (const float*)d_in[17];
  const float* ln2_b = (const float*)d_in[18];
  const float* fc_w = (const float*)d_in[19];
  const float* fc_b = (const float*)d_in[20];
  const float* proj_w = (const float*)d_in[21];
  const float* proj_b = (const float*)d_in[22];

  // ---- workspace layout (bump alloc, 256B aligned, with region reuse) -----
  char* ws = (char*)d_ws;
  size_t off = 0;
  auto take = [&](size_t bytes) -> char* {
    char* p = ws + off;
    off += (bytes + 255) & ~(size_t)255;
    return p;
  };
  f16* wqT = (f16*)take(512L * 768 * 2);
  f16* wkvT = (f16*)take(1024L * 768 * 2);
  f16* woT = (f16*)take(768L * 512 * 2);
  f16* inT = (f16*)take(2304L * 768 * 2);
  f16* outT = (f16*)take(768L * 768 * 2);
  f16* fcT = (f16*)take(3072L * 768 * 2);
  f16* projT = (f16*)take(768L * 3072 * 2);
  f16* ll = (f16*)take(512L * 768 * 2);     // LN(LN(latents)) f16
  f16* qf = (f16*)take(512L * 512 * 2);     // latent queries
  f16* kvl = (f16*)take(512L * 1024 * 2);   // latent k|v
  f16* o1 = (f16*)take(8192L * 512 * 2);    // perceiver attn out
  float* ybuf = (float*)take(8192L * 768 * 4);  // residual stream (L,bt,768)
  f16* zbuf = (f16*)take(8192L * 768 * 2);      // LN outputs
  char* regA = take(32768L * 768 * 2);   // xm, later MLP hidden (48MB each)
  char* regB = take(32768L * 1024 * 2);  // kvx (64MB), later qkv+o2 (50MB)
  f16* xm = (f16*)regA;
  f16* hbuf = (f16*)regA;
  f16* kvx = (f16*)regB;
  f16* qkv = (f16*)regB;
  f16* o2 = (f16*)(regB + 8192L * 2304 * 2);

  auto tgrid = [](long total) { return dim3((unsigned)((total + 255) / 256)); };

  // ---- 1) weights -> f16 transposed (N,K) ---------------------------------
  transpose_f32_f16<<<tgrid(768L * 512), 256, 0, stream>>>(Wq, wqT, 768, 512);
  transpose_f32_f16<<<tgrid(768L * 1024), 256, 0, stream>>>(Wkv, wkvT, 768, 1024);
  transpose_f32_f16<<<tgrid(512L * 768), 256, 0, stream>>>(Wo, woT, 512, 768);
  transpose_f32_f16<<<tgrid(768L * 2304), 256, 0, stream>>>(in_proj_w, inT, 768, 2304);
  transpose_f32_f16<<<tgrid(768L * 768), 256, 0, stream>>>(out_w, outT, 768, 768);
  transpose_f32_f16<<<tgrid(768L * 3072), 256, 0, stream>>>(fc_w, fcT, 768, 3072);
  transpose_f32_f16<<<tgrid(3072L * 768), 256, 0, stream>>>(proj_w, projT, 3072, 768);

  // ---- 2) layernorms ------------------------------------------------------
  ln768_kernel<<<512, 256, 0, stream>>>(latents, ll, nl0_g, nl0_b, pl_g, pl_b);
  ln768_kernel<<<32768, 256, 0, stream>>>(visual, xm, pm_g, pm_b, nullptr, nullptr);

  // ---- 3) perceiver projections (block tile 64x128) -----------------------
  gemm_wmma_kernel<<<dim3(4, 8), 128, 0, stream>>>(        // q = ll @ Wq
      ll, 768, wqT, 768, nullptr, nullptr, qf, 512, 768, GF_OUTF16);
  gemm_wmma_kernel<<<dim3(8, 512), 128, 0, stream>>>(      // kv_img = xm @ Wkv
      xm, 768, wkvT, 768, nullptr, nullptr, kvx, 1024, 768, GF_OUTF16);
  gemm_wmma_kernel<<<dim3(8, 8), 128, 0, stream>>>(        // kv_lat = ll @ Wkv
      ll, 768, wkvT, 768, nullptr, nullptr, kvl, 1024, 768, GF_OUTF16);

  // ---- 4) perceiver attention + output projection (remap -> (L,bt,768)) ---
  perceiver_attn_kernel<<<1024, 128, 0, stream>>>(qf, kvx, kvl, o1);
  gemm_wmma_kernel<<<dim3(6, 128), 128, 0, stream>>>(
      o1, 512, woT, 512, nullptr, nullptr, ybuf, 768, 512, GF_REMAP);

  // ---- 5) self-attention block -------------------------------------------
  ln768_kernel<<<8192, 256, 0, stream>>>(ybuf, zbuf, ln1_g, ln1_b, nullptr, nullptr);
  gemm_wmma_kernel<<<dim3(18, 128), 128, 0, stream>>>(     // qkv projection
      zbuf, 768, inT, 768, in_proj_b, nullptr, qkv, 2304, 768,
      GF_OUTF16 | GF_BIAS);
  self_attn_kernel<<<1536, 128, 0, stream>>>(qkv, o2);
  gemm_wmma_kernel<<<dim3(6, 128), 128, 0, stream>>>(      // y += o2@Wout + b
      o2, 768, outT, 768, out_b, ybuf, ybuf, 768, 768, GF_BIAS | GF_ADD);

  // ---- 6) MLP with QuickGELU ---------------------------------------------
  ln768_kernel<<<8192, 256, 0, stream>>>(ybuf, zbuf, ln2_g, ln2_b, nullptr, nullptr);
  gemm_wmma_kernel<<<dim3(24, 128), 128, 0, stream>>>(     // h = gelu(z@fc + b)
      zbuf, 768, fcT, 768, fc_b, nullptr, hbuf, 3072, 768,
      GF_OUTF16 | GF_BIAS | GF_GELU);
  gemm_wmma_kernel<<<dim3(6, 128), 128, 0, stream>>>(      // out = y + h@proj + b
      hbuf, 3072, projT, 3072, proj_b, ybuf, (float*)d_out, 768, 3072,
      GF_BIAS | GF_ADD);
}
